// AttnBlock_14826227106228
// MI455X (gfx1250) — matrix operand order repaired and checked by reading, not compiled, not run on hardware
//
#include <hip/hip_runtime.h>
#include <hip/hip_bf16.h>
#include <math.h>

// ---------------- problem constants (from reference) ----------------
#define S_   2048
#define E_   1024
#define H_   16
#define D_   64
#define NFC_ 4096
#define NEG_INF_ (-1e30f)
// MAX_LEN == S  =>  Er_s == Er (offset 0)

typedef __attribute__((ext_vector_type(16))) _Float16 v16h;
typedef __attribute__((ext_vector_type(8)))  float    v8f;

static __device__ __forceinline__ v8f wmma_f16(v16h a, v16h b, v8f c) {
  // D = A(16x32 f16) * B(32x16 f16) + C(16x16 f32)
  return __builtin_amdgcn_wmma_f32_16x16x32_f16(false, a, false, b, (short)0, c,
                                                false, false);
}

// ---- A fragment: 16x32 f16, row-major source, contiguous-K lane layout ----
// lane L<16 : row M=L, K = 0..15;  lane L>=16 : row M=L-16, K = 16..31
static __device__ __forceinline__ v16h load_A(const _Float16* src, int ld,
                                              int row0, int k0) {
  int lane = threadIdx.x & 31;
  int m    = lane & 15;
  int kb   = (lane >> 4) << 4;  // 0 or 16
  const _Float16* p = src + (size_t)(row0 + m) * ld + k0;
  v16h a;
#pragma unroll
  for (int g = 0; g < 2; ++g)
#pragma unroll
    for (int e = 0; e < 8; ++e)
      a[g * 8 + e] = p[kb + g * 8 + e];
  return a;
}

// ---- B fragment: 32x16 f16 from a K-contiguous (transposed) source ----
// element B[k][n] = src[(n0+n)*ld + kc + k];  lanes 0-15: K rows 0..15, lanes 16-31: 16..31
static __device__ __forceinline__ v16h load_BT(const _Float16* src, int ld,
                                               int kc, int n0) {
  int lane = threadIdx.x & 31;
  int n    = lane & 15;
  int kb   = (lane >> 4) << 4;  // 0 or 16
  const _Float16* p = src + (size_t)(n0 + n) * ld + kc + kb;
  v16h b;
#pragma unroll
  for (int v = 0; v < 16; ++v) b[v] = p[v];
  return b;
}

// same as load_BT but clamps the row index (Er band can run past the tensor)
static __device__ __forceinline__ v16h load_BT_clamp(const _Float16* src, int ld,
                                                     int kc, int n0, int maxrow) {
  int lane = threadIdx.x & 31;
  int n    = lane & 15;
  int kb   = (lane >> 4) << 4;
  int row  = n0 + n;
  row = row < 0 ? 0 : (row > maxrow ? maxrow : row);
  const _Float16* p = src + (size_t)row * ld + kc + kb;
  v16h b;
#pragma unroll
  for (int v = 0; v < 16; ++v) b[v] = p[v];
  return b;
}

// ---- B fragment: 32x16 f16 from an N-contiguous (row-major KxN) source ----
// element B[k][n] = src[(k0+k)*ld + n0 + n]
static __device__ __forceinline__ v16h load_B(const _Float16* src, int ld,
                                              int k0, int n0) {
  int lane = threadIdx.x & 31;
  int n    = lane & 15;
  int kb   = (lane >> 4) << 4;
  v16h b;
#pragma unroll
  for (int v = 0; v < 16; ++v)
    b[v] = src[(size_t)(k0 + kb + v) * ld + n0 + n];
  return b;
}

// ---------------- convert kernels ----------------
__global__ void cvt_t_kernel(const float* __restrict__ in, _Float16* __restrict__ out,
                             int Kd, int Nd) {  // in: KxN f32 -> out: NxK f16 (transposed)
  size_t t = (size_t)blockIdx.x * blockDim.x + threadIdx.x;
  if (t >= (size_t)Kd * Nd) return;
  int k = (int)(t / Nd), n = (int)(t % Nd);
  out[(size_t)n * Kd + k] = (_Float16)in[t];
}

__global__ void cvt_kernel(const float* __restrict__ in, _Float16* __restrict__ out,
                           int nElem) {
  int t = blockIdx.x * blockDim.x + threadIdx.x;
  if (t < nElem) out[t] = (_Float16)in[t];
}

// ---------------- LayerNorm (optionally fused residual add) ----------------
__global__ __launch_bounds__(256)
void ln_kernel(const float* __restrict__ in0, const float* __restrict__ in1,
               const float* __restrict__ gw, const float* __restrict__ gb,
               float* __restrict__ out32, _Float16* __restrict__ out16) {
  int row = blockIdx.x, tid = threadIdx.x;
  __shared__ float rsum[256], rsq[256];
  const float* p0 = in0 + (size_t)row * E_;
  const float* p1 = in1 ? in1 + (size_t)row * E_ : nullptr;
  float v[4], s = 0.f, sq = 0.f;
#pragma unroll
  for (int i = 0; i < 4; ++i) {
    int col = tid + i * 256;
    float x = p0[col];
    if (p1) x += p1[col];
    v[i] = x; s += x; sq += x * x;
  }
  rsum[tid] = s; rsq[tid] = sq;
  __syncthreads();
  for (int st = 128; st > 0; st >>= 1) {
    if (tid < st) { rsum[tid] += rsum[tid + st]; rsq[tid] += rsq[tid + st]; }
    __syncthreads();
  }
  float mean = rsum[0] * (1.f / E_);
  float var  = rsq[0] * (1.f / E_) - mean * mean;
  float rstd = rsqrtf(var + 1e-5f);
#pragma unroll
  for (int i = 0; i < 4; ++i) {
    int col = tid + i * 256;
    float y = (v[i] - mean) * rstd * gw[col] + gb[col];
    if (out32) out32[(size_t)row * E_ + col] = y;
    out16[(size_t)row * E_ + col] = (_Float16)y;
  }
}

// ---------------- generic WMMA GEMM: C = act(A @ Bt^T) (+ resid) ----------------
// A: MxK f16 row-major, Bt: NxK f16 (pre-transposed weight). 8 waves/block;
// each wave owns a 16x64 output strip (4 accumulators, A fragment reused 4x).
// Software-pipelined: next K-step's fragments are in flight during the WMMAs.
// Block covers 64 rows x 128 cols.  Requires N % 128 == 0, K % 32 == 0, K >= 64.
__global__ __launch_bounds__(256)
void gemm_kernel(const _Float16* __restrict__ A, const _Float16* __restrict__ Bt,
                 int M, int N, int K,
                 _Float16* __restrict__ C16, float* __restrict__ C32,
                 const float* __restrict__ resid, int act_gelu) {
  int wave = threadIdx.x >> 5;
  int lane = threadIdx.x & 31;
  int row0 = blockIdx.x * 64 + (wave >> 1) * 16;
  int colb = blockIdx.y * 128 + (wave & 1) * 64;
  (void)M;
  v8f acc[4] = {};

  // prologue: first fragment set
  v16h a  = load_A (A,  K, row0, 0);
  v16h b0 = load_BT(Bt, K, 0, colb);
  v16h b1 = load_BT(Bt, K, 0, colb + 16);
  v16h b2 = load_BT(Bt, K, 0, colb + 32);
  v16h b3 = load_BT(Bt, K, 0, colb + 48);

  for (int k0 = 0; k0 < K; k0 += 32) {
    // issue next K-step's loads (branchless: last iter re-reads a dead slab)
    int kn = k0 + 32 < K ? k0 + 32 : k0;
    __builtin_prefetch(A  + (size_t)(row0 + (lane & 15)) * K + kn + 64, 0, 1);
    __builtin_prefetch(Bt + (size_t)(colb + (lane & 15)) * K + kn + 64, 0, 1);
    v16h an  = load_A (A,  K, row0, kn);
    v16h c0  = load_BT(Bt, K, kn, colb);
    v16h c1  = load_BT(Bt, K, kn, colb + 16);
    v16h c2  = load_BT(Bt, K, kn, colb + 32);
    v16h c3  = load_BT(Bt, K, kn, colb + 48);

    acc[0] = wmma_f16(a, b0, acc[0]);
    acc[1] = wmma_f16(a, b1, acc[1]);
    acc[2] = wmma_f16(a, b2, acc[2]);
    acc[3] = wmma_f16(a, b3, acc[3]);

    a = an; b0 = c0; b1 = c1; b2 = c2; b3 = c3;
  }

  int mhi = (lane >> 4) * 8;
  int n   = lane & 15;
#pragma unroll
  for (int t = 0; t < 4; ++t)
#pragma unroll
    for (int r = 0; r < 8; ++r) {
      int m = r + mhi;
      size_t idx = (size_t)(row0 + m) * N + colb + 16 * t + n;
      float v = acc[t][r];
      if (act_gelu) v = 0.5f * v * (1.0f + erff(v * 0.70710678118f));
      if (resid) v += resid[idx];
      if (C32) C32[idx] = v;
      if (C16) C16[idx] = (_Float16)v;
    }
}

// ---------------- flash attention with relative-position skew ----------------
// one wave per (head, 16-query tile). 32-key tiles.
// bias[q,k] = (Q[q] . Er[S-1-q+k]) handled via a 16x48 WMMA band + LDS diagonal gather.
__global__ __launch_bounds__(32)
void attn_kernel(const _Float16* __restrict__ Q, const _Float16* __restrict__ K,
                 const _Float16* __restrict__ V, const _Float16* __restrict__ Er,
                 const int* __restrict__ kpm, float* __restrict__ Out) {
  int qt   = blockIdx.x & ((S_ / 16) - 1);   // 128 q-tiles
  int head = blockIdx.x >> 7;
  int q0   = qt * 16;
  int lane = threadIdx.x & 31;
  int hi   = lane >> 4;
  int n    = lane & 15;

  __shared__ float    P2[16][48];   // Q.Er^T band
  __shared__ _Float16 Pp[16][32];   // probabilities, re-laid out for A fragment

  const _Float16* Qh = Q + head * D_;
  const _Float16* Kh = K + head * D_;
  const _Float16* Vh = V + head * D_;

  v16h aq0 = load_A(Qh, E_, q0, 0);
  v16h aq1 = load_A(Qh, E_, q0, 32);

  float rm[8], rl[8];
  v8f   o[4] = {};
#pragma unroll
  for (int r = 0; r < 8; ++r) { rm[r] = -1e30f; rl[r] = 0.f; }

  int ktiles = (q0 + 16 + 31) >> 5;
  for (int kt = 0; kt < ktiles; ++kt) {
    int k0 = kt * 32;
    if (kt + 1 < ktiles) {  // prefetch next K/V tiles
      __builtin_prefetch(Kh + (size_t)(k0 + 32 + n) * E_, 0, 1);
      __builtin_prefetch(Vh + (size_t)(k0 + 32 + n) * E_, 0, 1);
    }

    // ---- scores S = Q K^T (16x32 over two 16x16 accumulators) ----
    v8f sa = {}, sb = {};
    sa = wmma_f16(aq0, load_BT(Kh, E_, 0,  k0),      sa);
    sa = wmma_f16(aq1, load_BT(Kh, E_, 32, k0),      sa);
    sb = wmma_f16(aq0, load_BT(Kh, E_, 0,  k0 + 16), sb);
    sb = wmma_f16(aq1, load_BT(Kh, E_, 32, k0 + 16), sb);

    // ---- relative-position band: P2[m][c] = Q[q0+m] . Er[jb+c] ----
    int jb = S_ - 1 - q0 + k0 - 15;  // c = 15 + ki - qi  maps bias diag
#pragma unroll
    for (int t = 0; t < 3; ++t) {
      v8f pe = {};
      pe = wmma_f16(aq0, load_BT_clamp(Er, D_, 0,  jb + 16 * t, S_ - 1), pe);
      pe = wmma_f16(aq1, load_BT_clamp(Er, D_, 32, jb + 16 * t, S_ - 1), pe);
#pragma unroll
      for (int r = 0; r < 8; ++r) P2[r + 8 * hi][16 * t + n] = pe[r];
    }
    __syncthreads();

    // ---- mask + online softmax (rows live across 16 lanes per VGPR) ----
    float pa[8], pb[8];
#pragma unroll
    for (int r = 0; r < 8; ++r) {
      int m   = r + 8 * hi;
      int kia = k0 + n, kib = k0 + 16 + n;
      float va = (sa[r] + P2[m][15 + n - m]) * 0.125f;  // 1/sqrt(64)
      float vb = (sb[r] + P2[m][31 + n - m]) * 0.125f;
      bool oka = (kia <= q0 + m) && (kpm[kia] != 0);
      bool okb = (kib <= q0 + m) && (kpm[kib] != 0);
      va = oka ? va : NEG_INF_;
      vb = okb ? vb : NEG_INF_;
      float mx = fmaxf(va, vb);
      for (int off = 1; off < 16; off <<= 1)
        mx = fmaxf(mx, __shfl_xor(mx, off, 32));
      float mnew  = fmaxf(rm[r], mx);
      float alpha = __expf(rm[r] - mnew);
      float ea = oka ? __expf(va - mnew) : 0.f;
      float eb = okb ? __expf(vb - mnew) : 0.f;
      float rs = ea + eb;
      for (int off = 1; off < 16; off <<= 1)
        rs += __shfl_xor(rs, off, 32);
      rl[r] = rl[r] * alpha + rs;
      rm[r] = mnew;
      pa[r] = ea; pb[r] = eb;
#pragma unroll
      for (int c = 0; c < 4; ++c) o[c][r] *= alpha;
    }

    // ---- stage probs to LDS, reload as A fragment, O += P @ V ----
#pragma unroll
    for (int r = 0; r < 8; ++r) {
      int m = r + 8 * hi;
      Pp[m][n]      = (_Float16)pa[r];
      Pp[m][16 + n] = (_Float16)pb[r];
    }
    __syncthreads();
    v16h ap = load_A(&Pp[0][0], 32, 0, 0);
#pragma unroll
    for (int c = 0; c < 4; ++c)
      o[c] = wmma_f16(ap, load_B(Vh + c * 16, E_, k0, 0), o[c]);
    __syncthreads();
  }

  // ---- epilogue: normalize and write (S, E) attention output ----
#pragma unroll
  for (int c = 0; c < 4; ++c)
#pragma unroll
    for (int r = 0; r < 8; ++r) {
      int m = r + 8 * hi;
      float inv = 1.f / fmaxf(rl[r], 1e-20f);
      Out[(size_t)(q0 + m) * E_ + head * D_ + c * 16 + n] = o[c][r] * inv;
    }
}

// ---------------- host side ----------------
static size_t alignup(size_t x) { return (x + 255) & ~(size_t)255; }

extern "C" void kernel_launch(void* const* d_in, const int* in_sizes, int n_in,
                              void* d_out, int out_size, void* d_ws, size_t ws_size,
                              hipStream_t stream) {
  (void)in_sizes; (void)n_in; (void)out_size; (void)ws_size;
  const float* x      = (const float*)d_in[0];
  const int*   kpm    = (const int*)  d_in[1];
  const float* Wq     = (const float*)d_in[2];
  const float* Wk     = (const float*)d_in[3];
  const float* Wv     = (const float*)d_in[4];
  const float* Er     = (const float*)d_in[5];
  const float* ln1w   = (const float*)d_in[6];
  const float* ln1b   = (const float*)d_in[7];
  const float* ln2w   = (const float*)d_in[8];
  const float* ln2b   = (const float*)d_in[9];
  const float* fc_w   = (const float*)d_in[10];
  const float* proj_w = (const float*)d_in[11];
  float* out = (float*)d_out;

  // workspace carve-up
  char* ws = (char*)d_ws;
  size_t off = 0;
  float*    h     = (float*)   (ws + off); off = alignup(off + (size_t)S_*E_*4);
  _Float16* h16   = (_Float16*)(ws + off); off = alignup(off + (size_t)S_*E_*2);
  _Float16* WqT   = (_Float16*)(ws + off); off = alignup(off + (size_t)E_*E_*2);
  _Float16* WkT   = (_Float16*)(ws + off); off = alignup(off + (size_t)E_*E_*2);
  _Float16* WvT   = (_Float16*)(ws + off); off = alignup(off + (size_t)E_*E_*2);
  _Float16* Er16  = (_Float16*)(ws + off); off = alignup(off + (size_t)S_*D_*2);
  _Float16* q16   = (_Float16*)(ws + off); off = alignup(off + (size_t)S_*E_*2);
  _Float16* k16   = (_Float16*)(ws + off); off = alignup(off + (size_t)S_*E_*2);
  _Float16* v16   = (_Float16*)(ws + off); off = alignup(off + (size_t)S_*E_*2);
  float*    attn  = (float*)   (ws + off); off = alignup(off + (size_t)S_*E_*4);
  float*    h2    = (float*)   (ws + off); off = alignup(off + (size_t)S_*E_*4);
  _Float16* m16   = (_Float16*)(ws + off); off = alignup(off + (size_t)S_*E_*2);
  _Float16* fcT   = (_Float16*)(ws + off); off = alignup(off + (size_t)E_*NFC_*2);
  _Float16* g16   = (_Float16*)(ws + off); off = alignup(off + (size_t)S_*NFC_*2);
  _Float16* projT = (_Float16*)(ws + off); off = alignup(off + (size_t)NFC_*E_*2);

  // 1) convert + transpose weights to f16
  {
    int nEE = E_ * E_;
    cvt_t_kernel<<<(nEE + 255) / 256, 256, 0, stream>>>(Wq, WqT, E_, E_);
    cvt_t_kernel<<<(nEE + 255) / 256, 256, 0, stream>>>(Wk, WkT, E_, E_);
    cvt_t_kernel<<<(nEE + 255) / 256, 256, 0, stream>>>(Wv, WvT, E_, E_);
    int nFC = E_ * NFC_;
    cvt_t_kernel<<<(nFC + 255) / 256, 256, 0, stream>>>(fc_w, fcT, E_, NFC_);
    cvt_t_kernel<<<(nFC + 255) / 256, 256, 0, stream>>>(proj_w, projT, NFC_, E_);
    int nEr = S_ * D_;
    cvt_kernel<<<(nEr + 255) / 256, 256, 0, stream>>>(Er, Er16, nEr);
  }

  // 2) h = LN1(x)   (f32 for residual, f16 for GEMMs)
  ln_kernel<<<S_, 256, 0, stream>>>(x, nullptr, ln1w, ln1b, h, h16);

  // 3) Q,K,V = h @ W{q,k,v}
  dim3 gQKV(S_ / 64, E_ / 128);
  gemm_kernel<<<gQKV, 256, 0, stream>>>(h16, WqT, S_, E_, E_, q16, nullptr, nullptr, 0);
  gemm_kernel<<<gQKV, 256, 0, stream>>>(h16, WkT, S_, E_, E_, k16, nullptr, nullptr, 0);
  gemm_kernel<<<gQKV, 256, 0, stream>>>(h16, WvT, S_, E_, E_, v16, nullptr, nullptr, 0);

  // 4) causal rel-pos attention
  attn_kernel<<<(S_ / 16) * H_, 32, 0, stream>>>(q16, k16, v16, Er16, kpm, attn);

  // 5) h2 = LN2(h + attn) ; m = f16 copy of h2
  ln_kernel<<<S_, 256, 0, stream>>>(h, attn, ln2w, ln2b, h2, m16);

  // 6) g = gelu(m @ fc_w)
  dim3 gFC(S_ / 64, NFC_ / 128);
  gemm_kernel<<<gFC, 256, 0, stream>>>(m16, fcT, S_, NFC_, E_, g16, nullptr, nullptr, 1);

  // 7) out = h2 + g @ proj_w
  dim3 gPR(S_ / 64, E_ / 128);
  gemm_kernel<<<gPR, 256, 0, stream>>>(g16, projT, S_, E_, NFC_, nullptr, out, h2, 0);
}
